// LanguageModel_85237920957023
// MI455X (gfx1250) — compile-verified, matrix operand order checked by reading, not run on hardware
//
#include <hip/hip_runtime.h>
#include <hip/hip_bf16.h>

// ---------------------------------------------------------------------------
// Types for gfx1250 WMMA (wave32): v_wmma_f32_16x16x32_bf16
// ---------------------------------------------------------------------------
typedef __attribute__((ext_vector_type(16))) __bf16 v16bf;
typedef __attribute__((ext_vector_type(8)))  __bf16 v8bf;
typedef __attribute__((ext_vector_type(8)))  float  v8f;

#define D_EMB  1024
#define D_RNN  1024
#define NB     16      // batch
#define NT     256     // timesteps
#define NG     4096    // 4*D_RNN
#define NV     32000   // vocab
#define NBLK   64      // blocks in persistent recurrence kernel
#define FLAG_STRIDE 16 // one 64B cacheline per barrier flag

// ---------------------------------------------------------------------------
// WMMA helpers
// ---------------------------------------------------------------------------
__device__ __forceinline__ v8f wmma_bf16(v16bf a, v16bf b, v8f c) {
    // (neg_a, A, neg_b, B, c_mod, C, reuse_a, reuse_b)
    return __builtin_amdgcn_wmma_f32_16x16x32_bf16(false, a, false, b,
                                                   (short)0, c, false, false);
}

// Load one 16x32 bf16 fragment for the current lane.
// Layout (ISA 7.12.2, 16-bit A 16x32): lane L holds row M=L&15,
//   VGPR0..3 -> K = kg+0..kg+7, VGPR4..7 -> K = 16+kg..16+kg+7, kg = 8*(L>=16).
// Both runs are 8 contiguous bf16 = one 16B load each.
// B fragments use the symmetric layout with N in place of M, so B matrices are
// stored pre-transposed as [N][K] and loaded with the same helper.
__device__ __forceinline__ v16bf load_frag(const __bf16* base, int stride, int lane) {
    int m  = lane & 15;
    int kg = (lane >> 4) << 3;
    const __bf16* p = base + (size_t)m * stride + kg;
    v8bf lo = *(const v8bf*)(p);
    v8bf hi = *(const v8bf*)(p + 16);
    return __builtin_shufflevector(lo, hi, 0,1,2,3,4,5,6,7,8,9,10,11,12,13,14,15);
}

__device__ __forceinline__ float sigmoidf_(float x) {
    return 1.0f / (1.0f + __expf(-x));
}

// [16,1024] @ Bt[n0..n0+15][1024]^T -> one 16x16 tile, K=1024.
// Two interleaved accumulators halve the WMMA->WMMA RAW chain.
__device__ __forceinline__ v8f gemm_tile(const __bf16* __restrict__ h,
                                         const __bf16* __restrict__ Bt,
                                         int n0, int lane, v8f acc) {
    const __bf16* brow = Bt + (size_t)n0 * 1024;
    v8f a0 = {}; v8f a1 = {};
#pragma unroll 4
    for (int kb = 0; kb < 1024; kb += 64) {
        a0 = wmma_bf16(load_frag(h + kb, 1024, lane),
                       load_frag(brow + kb, 1024, lane), a0);
        a1 = wmma_bf16(load_frag(h + kb + 32, 1024, lane),
                       load_frag(brow + kb + 32, 1024, lane), a1);
    }
#pragma unroll
    for (int r = 0; r < 8; ++r) acc[r] += a0[r] + a1[r];
    return acc;
}

// ---------------------------------------------------------------------------
// fp32 [K][N] -> bf16 [N][K] transpose+convert (LDS tiled, 32x32)
// ---------------------------------------------------------------------------
__global__ __launch_bounds__(256) void transpose_convert(
    const float* __restrict__ in, __bf16* __restrict__ out, int K, int N)
{
    __shared__ float tile[32][33];
    int n0 = blockIdx.x * 32, k0 = blockIdx.y * 32;
    int tx = threadIdx.x & 31, ty = threadIdx.x >> 5;   // 8 rows / pass
#pragma unroll
    for (int p = 0; p < 4; ++p) {
        int r = ty + p * 8;
        tile[r][tx] = in[(size_t)(k0 + r) * N + n0 + tx];
    }
    __syncthreads();
#pragma unroll
    for (int p = 0; p < 4; ++p) {
        int r = ty + p * 8;
        out[(size_t)(n0 + r) * K + k0 + tx] = (__bf16)tile[tx][r];
    }
}

// ---------------------------------------------------------------------------
// Embedding gather + sqrt(D_EMB)=32 scale -> bf16 X, row = t*NB + b
// ---------------------------------------------------------------------------
__global__ __launch_bounds__(256) void gather_embed(
    const int* __restrict__ seq, const float* __restrict__ emb,
    __bf16* __restrict__ X)
{
    int row = blockIdx.x;            // t*NB + b
    int t = row >> 4, b = row & 15;
    int tok = seq[b * NT + t];
    int d = threadIdx.x * 4;
    float4 v = *(const float4*)(emb + (size_t)tok * D_EMB + d);
    __bf16* o = X + (size_t)row * D_EMB + d;
    o[0] = (__bf16)(v.x * 32.0f);
    o[1] = (__bf16)(v.y * 32.0f);
    o[2] = (__bf16)(v.z * 32.0f);
    o[3] = (__bf16)(v.w * 32.0f);
}

__global__ __launch_bounds__(256) void init_flags(unsigned* flags) {
    flags[blockIdx.x * 256 + threadIdx.x] = 0u;   // NBLK*FLAG_STRIDE entries
}

// ---------------------------------------------------------------------------
// Big GEMM: C[M,N] = A[M,K=1024] * Bt[N,K]^T.
// 64(M) x 128(N) block tile, 8 waves of 32x32, double-buffered LDS with a
// one-sync-per-chunk software pipeline (global loads overlap WMMA).
// MODE 0: plain fp32 store (used for X@W0 -> Z0all)
// MODE 1: out[(b*NT+t)*N + n] = acc + bias[n], row = t*NB+b (output logits)
// ---------------------------------------------------------------------------
template <int MODE>
__global__ __launch_bounds__(256) void gemm_bf16_64x128(
    const __bf16* __restrict__ A, const __bf16* __restrict__ Bt,
    float* __restrict__ C, const float* __restrict__ bias, int M, int N)
{
    constexpr int K  = 1024;
    constexpr int CH = K / 32;                 // 32 K-chunks
    __shared__ __bf16 As[2][64 * 40];
    __shared__ __bf16 Bs[2][128 * 40];

    int tid  = threadIdx.x;
    int lane = tid & 31, w = tid >> 5;
    int m0 = blockIdx.y * 64, n0 = blockIdx.x * 128;
    int mw = (w >> 2) * 32, nw = (w & 3) * 32;

    int ar = (tid * 8) >> 5,  ac = (tid * 8) & 31;    // A: 8 bf16/thread
    int br = (tid * 16) >> 5, bc = (tid * 16) & 31;   // B: 16 bf16/thread

    const __bf16* Ap = A  + (size_t)(m0 + ar) * K + ac;
    const __bf16* Bp = Bt + (size_t)(n0 + br) * K + bc;

    v8f acc00 = {}, acc01 = {}, acc10 = {}, acc11 = {};

    // prologue: fill buffer 0
    v8bf va  = *(const v8bf*)(Ap);
    v8bf vb0 = *(const v8bf*)(Bp);
    v8bf vb1 = *(const v8bf*)(Bp + 8);
    *(v8bf*)(&As[0][ar * 40 + ac])     = va;
    *(v8bf*)(&Bs[0][br * 40 + bc])     = vb0;
    *(v8bf*)(&Bs[0][br * 40 + bc + 8]) = vb1;
    __syncthreads();

#pragma unroll 2
    for (int k = 0; k < CH; ++k) {
        int cur = k & 1;
        if (k + 1 < CH) {                       // issue next chunk's loads
            int kb = (k + 1) * 32;
            va  = *(const v8bf*)(Ap + kb);
            vb0 = *(const v8bf*)(Bp + kb);
            vb1 = *(const v8bf*)(Bp + kb + 8);
        }
        v16bf fa0 = load_frag(&As[cur][mw * 40], 40, lane);
        v16bf fa1 = load_frag(&As[cur][(mw + 16) * 40], 40, lane);
        v16bf fb0 = load_frag(&Bs[cur][nw * 40], 40, lane);
        v16bf fb1 = load_frag(&Bs[cur][(nw + 16) * 40], 40, lane);
        acc00 = wmma_bf16(fa0, fb0, acc00);
        acc01 = wmma_bf16(fa0, fb1, acc01);
        acc10 = wmma_bf16(fa1, fb0, acc10);
        acc11 = wmma_bf16(fa1, fb1, acc11);
        if (k + 1 < CH) {                       // stage next chunk
            int nxt = cur ^ 1;
            *(v8bf*)(&As[nxt][ar * 40 + ac])     = va;
            *(v8bf*)(&Bs[nxt][br * 40 + bc])     = vb0;
            *(v8bf*)(&Bs[nxt][br * 40 + bc + 8]) = vb1;
        }
        __syncthreads();
    }

    int hi8 = (lane >> 4) << 3;
    int col = lane & 15;
#pragma unroll
    for (int r = 0; r < 8; ++r) {
        float v00 = acc00[r], v01 = acc01[r], v10 = acc10[r], v11 = acc11[r];
        int gm0 = m0 + mw + hi8 + r;
        int gm1 = gm0 + 16;
        int gn0 = n0 + nw + col;
        int gn1 = gn0 + 16;
        if (MODE == 0) {
            C[(size_t)gm0 * N + gn0] = v00;
            C[(size_t)gm0 * N + gn1] = v01;
            C[(size_t)gm1 * N + gn0] = v10;
            C[(size_t)gm1 * N + gn1] = v11;
        } else {
            int t0 = gm0 >> 4, b0i = gm0 & 15;          // row = t*NB + b
            int t1 = gm1 >> 4, b1i = gm1 & 15;
            size_t base0 = ((size_t)b0i * NT + t0) * (size_t)N;
            size_t base1 = ((size_t)b1i * NT + t1) * (size_t)N;
            float bi0 = bias[gn0], bi1 = bias[gn1];
            C[base0 + gn0] = v00 + bi0;
            C[base0 + gn1] = v01 + bi1;
            C[base1 + gn0] = v10 + bi0;
            C[base1 + gn1] = v11 + bi1;
        }
    }
}

// ---------------------------------------------------------------------------
// Distributed global barrier: each block releases its own 64B flag (parallel
// L2 stores); 64 threads spin one-flag-each (one gathered load per wave per
// poll). Agent-scope release/acquire fences flush to L2 / invalidate WGP$.
// ---------------------------------------------------------------------------
__device__ __forceinline__ void global_barrier(unsigned* flags, unsigned phase) {
    __syncthreads();
    if (threadIdx.x == 0) {
        __builtin_amdgcn_fence(__ATOMIC_RELEASE, "agent");
        __hip_atomic_store(&flags[blockIdx.x * FLAG_STRIDE], phase,
                           __ATOMIC_RELAXED, __HIP_MEMORY_SCOPE_AGENT);
    }
    if (threadIdx.x < NBLK) {
        while (__hip_atomic_load(&flags[threadIdx.x * FLAG_STRIDE],
                                 __ATOMIC_RELAXED,
                                 __HIP_MEMORY_SCOPE_AGENT) < phase) {
            __builtin_amdgcn_s_sleep(1);
        }
    }
    __syncthreads();
    __builtin_amdgcn_fence(__ATOMIC_ACQUIRE, "agent");
}

// ---------------------------------------------------------------------------
// Persistent LSTM recurrence: 64 blocks x 128 threads, all 256 steps in one
// launch. Block bi owns gate columns j in [16*bi, 16*bi+16); wave w computes
// the 16x16 tile of gate w (n = w*1024 + j0). Cell state lives in LDS.
// ---------------------------------------------------------------------------
__global__ __launch_bounds__(128) void lstm_persistent(
    const __bf16* __restrict__ U0t, const __bf16* __restrict__ W1t,
    const __bf16* __restrict__ U1t, const float* __restrict__ Z0all,
    const float* __restrict__ b0,  const float* __restrict__ b1,
    __bf16* __restrict__ h0b, __bf16* __restrict__ h1b,
    __bf16* __restrict__ H1all, unsigned* __restrict__ flags)
{
    __shared__ float zbuf[4][16][16];
    __shared__ float c0l[16][16];
    __shared__ float c1l[16][16];

    int tid  = threadIdx.x;
    int lane = tid & 31, w = tid >> 5;
    int j0   = blockIdx.x * 16;
    int hi8  = (lane >> 4) << 3;
    int jc   = lane & 15;
    int n    = w * 1024 + j0 + jc;          // this wave's gate column (global)
    unsigned phase = 0;

    // Zero cell state (LDS) and this block's h slices (global).
    for (int e = tid; e < 256; e += 128) {
        int m = e >> 4, j = e & 15;
        c0l[m][j] = 0.0f;
        c1l[m][j] = 0.0f;
        h0b[m * 1024 + j0 + j] = (__bf16)0.0f;
        h1b[m * 1024 + j0 + j] = (__bf16)0.0f;
    }
    global_barrier(flags, ++phase);

    for (int t = 0; t < NT; ++t) {
        // ---- Layer 0: z0 = xW0[t] + h0 @ U0^T + b0 ----
        {
            v8f acc = {};
            acc = gemm_tile(h0b, U0t, w * 1024 + j0, lane, acc);
            const float* zrow = Z0all + (size_t)t * NB * NG;
            float bias = b0[n];
#pragma unroll
            for (int r = 0; r < 8; ++r) {
                int m = hi8 + r;
                zbuf[w][m][jc] = acc[r] + zrow[(size_t)m * NG + n] + bias;
            }
        }
        __syncthreads();
        for (int e = tid; e < 256; e += 128) {
            int m = e >> 4, j = e & 15;
            float gi = zbuf[0][m][j], gf = zbuf[1][m][j];
            float gg = zbuf[2][m][j], go = zbuf[3][m][j];
            float cn = sigmoidf_(gf) * c0l[m][j] + sigmoidf_(gi) * tanhf(gg);
            c0l[m][j] = cn;
            h0b[m * 1024 + j0 + j] = (__bf16)(sigmoidf_(go) * tanhf(cn));
        }
        global_barrier(flags, ++phase);

        // ---- Layer 1: z1 = h0 @ W1^T + h1 @ U1^T + b1 ----
        {
            v8f acc = {};
            acc = gemm_tile(h0b, W1t, w * 1024 + j0, lane, acc);
            acc = gemm_tile(h1b, U1t, w * 1024 + j0, lane, acc);
            float bias = b1[n];
#pragma unroll
            for (int r = 0; r < 8; ++r) {
                int m = hi8 + r;
                zbuf[w][m][jc] = acc[r] + bias;
            }
        }
        __syncthreads();
        for (int e = tid; e < 256; e += 128) {
            int m = e >> 4, j = e & 15;
            float gi = zbuf[0][m][j], gf = zbuf[1][m][j];
            float gg = zbuf[2][m][j], go = zbuf[3][m][j];
            float cn = sigmoidf_(gf) * c1l[m][j] + sigmoidf_(gi) * tanhf(gg);
            c1l[m][j] = cn;
            __bf16 hv = (__bf16)(sigmoidf_(go) * tanhf(cn));
            h1b[m * 1024 + j0 + j] = hv;
            H1all[((size_t)t * NB + m) * D_RNN + j0 + j] = hv;
        }
        global_barrier(flags, ++phase);
    }
}

// ---------------------------------------------------------------------------
// Host launcher
// ---------------------------------------------------------------------------
extern "C" void kernel_launch(void* const* d_in, const int* in_sizes, int n_in,
                              void* d_out, int out_size, void* d_ws, size_t ws_size,
                              hipStream_t stream)
{
    (void)in_sizes; (void)n_in; (void)out_size; (void)ws_size;

    const int*   seq  = (const int*)  d_in[0];
    const float* emb  = (const float*)d_in[1];
    const float* W0   = (const float*)d_in[2];
    const float* U0   = (const float*)d_in[3];
    const float* b0   = (const float*)d_in[4];
    const float* W1   = (const float*)d_in[5];
    const float* U1   = (const float*)d_in[6];
    const float* b1   = (const float*)d_in[7];
    const float* Wout = (const float*)d_in[8];
    const float* bout = (const float*)d_in[9];
    float* out = (float*)d_out;

    char* ws = (char*)d_ws;
    size_t off = 0;
    auto alloc = [&](size_t bytes) -> void* {
        void* p = ws + off;
        off += (bytes + 255) & ~(size_t)255;
        return p;
    };

    __bf16*   W0t   = (__bf16*)  alloc((size_t)NG * 1024 * 2);
    __bf16*   U0t   = (__bf16*)  alloc((size_t)NG * 1024 * 2);
    __bf16*   W1t   = (__bf16*)  alloc((size_t)NG * 1024 * 2);
    __bf16*   U1t   = (__bf16*)  alloc((size_t)NG * 1024 * 2);
    __bf16*   Woutt = (__bf16*)  alloc((size_t)NV * 1024 * 2);
    __bf16*   Xb    = (__bf16*)  alloc((size_t)NT * NB * 1024 * 2);
    float*    Z0all = (float*)   alloc((size_t)NT * NB * NG * 4);
    __bf16*   H1all = (__bf16*)  alloc((size_t)NT * NB * 1024 * 2);
    __bf16*   h0b   = (__bf16*)  alloc((size_t)NB * D_RNN * 2);
    __bf16*   h1b   = (__bf16*)  alloc((size_t)NB * D_RNN * 2);
    unsigned* flags = (unsigned*)alloc((size_t)NBLK * FLAG_STRIDE * 4);

    // 1) Weight conversion: fp32 [K][N] -> bf16 [N][K]
    transpose_convert<<<dim3(NG / 32, 1024 / 32), 256, 0, stream>>>(W0,   W0t,   1024, NG);
    transpose_convert<<<dim3(NG / 32, 1024 / 32), 256, 0, stream>>>(U0,   U0t,   1024, NG);
    transpose_convert<<<dim3(NG / 32, 1024 / 32), 256, 0, stream>>>(W1,   W1t,   1024, NG);
    transpose_convert<<<dim3(NG / 32, 1024 / 32), 256, 0, stream>>>(U1,   U1t,   1024, NG);
    transpose_convert<<<dim3(NV / 32, 1024 / 32), 256, 0, stream>>>(Wout, Woutt, 1024, NV);

    // 2) Embedding gather + scale; zero barrier flags
    gather_embed<<<NT * NB, 256, 0, stream>>>(seq, emb, Xb);
    init_flags<<<(NBLK * FLAG_STRIDE) / 256, 256, 0, stream>>>(flags);

    // 3) Precompute X@W0 for all timesteps (parallel over T)
    gemm_bf16_64x128<0><<<dim3(NG / 128, (NT * NB) / 64), 256, 0, stream>>>(
        Xb, W0t, Z0all, nullptr, NT * NB, NG);

    // 4) Entire 256-step recurrence in ONE persistent kernel
    lstm_persistent<<<NBLK, 128, 0, stream>>>(
        U0t, W1t, U1t, Z0all, b0, b1, h0b, h1b, H1all, flags);

    // 5) Output projection for all timesteps: H1 @ Wout + bout -> [B,T,V]
    gemm_bf16_64x128<1><<<dim3(NV / 128, (NT * NB) / 64), 256, 0, stream>>>(
        H1all, Woutt, out, bout, NT * NB, NV);
}
